// MAGAT_33268816674925
// MI455X (gfx1250) — matrix operand order repaired and checked by reading, not compiled, not run on hardware
//
#include <hip/hip_runtime.h>

typedef unsigned int u32;
typedef __attribute__((ext_vector_type(16))) __bf16 v16bf;
typedef __attribute__((ext_vector_type(8)))  float  v8f;

#define N_ 4096
#define F_ 128
#define D_ 128
#define H_ 4
#define ALPHA_ 0.2f
#define LOG2E_ 1.4426950408889634f

union FragBF {
    v16bf v;
    u32   u[8];
};

// pack two f32 into a dword of bf16 (truncation): single v_perm_b32
__device__ __forceinline__ u32 pack_bf16(float a, float b) {
    return __builtin_amdgcn_perm(__float_as_uint(b), __float_as_uint(a), 0x07060302u);
}

__device__ __forceinline__ float elu_f(float x) {
    return x > 0.f ? x : (__expf(x) - 1.f);
}

// ---------------------------------------------------------------------------
// wa[h][f] = sum_d W[h][f][d] * a[h][d]  (reassociation: e = x0 . (W @ a))
// ---------------------------------------------------------------------------
__global__ __launch_bounds__(128) void wa_kernel(
    const float* __restrict__ W, const float* __restrict__ a_src,
    const float* __restrict__ a_dst, float* __restrict__ wa_src,
    float* __restrict__ wa_dst) {
    int t = blockIdx.x * blockDim.x + threadIdx.x;  // 0..H*F-1
    int h = t >> 7;
    const float* wr = W + (size_t)t * D_;
    const float* as = a_src + h * D_;
    const float* ad = a_dst + h * D_;
    float ss = 0.f, sd = 0.f;
    #pragma unroll 8
    for (int d = 0; d < D_; ++d) {
        float w = wr[d];
        ss += w * as[d];
        sd += w * ad[d];
    }
    wa_src[t] = ss;
    wa_dst[t] = sd;
}

// e values pre-scaled by log2(e) so the attention loop uses raw v_exp_f32.
__global__ __launch_bounds__(128) void e_kernel(
    const float* __restrict__ x0, const float* __restrict__ wa_src,
    const float* __restrict__ wa_dst, float* __restrict__ e_src,
    float* __restrict__ e_dst) {
    int t = blockIdx.x * blockDim.x + threadIdx.x;  // 0..H*N-1
    int h = t >> 12;
    int n = t & (N_ - 1);
    const float* xr = x0 + (size_t)n * F_;
    const float* ws = wa_src + h * F_;
    const float* wd = wa_dst + h * F_;
    float ss = 0.f, sd = 0.f;
    #pragma unroll 8
    for (int f = 0; f < F_; ++f) {
        float x = xr[f];
        ss += x * ws[f];
        sd += x * wd[f];
    }
    e_src[t] = ss * LOG2E_;
    e_dst[t] = sd * LOG2E_;
}

// ---------------------------------------------------------------------------
// Wh = x0 @ W[h]  (bf16 WMMA) -> WMMA-B-fragment swizzled global layout:
//   WhB dword = ((((h*128 + kblock)*8 + g)*32 + flane)*8 + v)
//   flane = khalf*16 + n, dword v packs nodes (kblock*32 + khalf*16 + 2v, +1)
// ---------------------------------------------------------------------------
__global__ __launch_bounds__(256) void wh_kernel(
    const float* __restrict__ x0, const float* __restrict__ W,
    u32* __restrict__ whB) {
    __shared__ __align__(16) u32 xs[16 * 64];  // 16 x 128 bf16, pair-packed
    const int h   = blockIdx.y;
    const int r0  = blockIdx.x * 16;
    const int tid = threadIdx.x;

    {   // cooperative x0 tile load + bf16 convert
        int mrow = tid >> 4;
        int c8   = (tid & 15) * 8;
        const float* src = x0 + (size_t)(r0 + mrow) * F_ + c8;
        float4 f0 = *(const float4*)(src);
        float4 f1 = *(const float4*)(src + 4);
        uint4 q;
        q.x = pack_bf16(f0.x, f0.y);
        q.y = pack_bf16(f0.z, f0.w);
        q.z = pack_bf16(f1.x, f1.y);
        q.w = pack_bf16(f1.z, f1.w);
        *(uint4*)&xs[mrow * 64 + (tid & 15) * 4] = q;
    }
    __syncthreads();

    const int wave = tid >> 5;    // d-group g
    const int lane = tid & 31;
    const int n    = lane & 15;
    const int kh   = lane >> 4;
    const int d    = wave * 16 + n;
    const float* Wp = W + (size_t)h * F_ * D_;

    v8f acc = {};
    #pragma unroll
    for (int kb = 0; kb < 4; ++kb) {
        FragBF A, B;
        const u32* row = &xs[n * 64 + kb * 16 + kh * 4];
        *(uint4*)&A.u[0] = *(const uint4*)(row);
        *(uint4*)&A.u[4] = *(const uint4*)(row + 8);
        #pragma unroll
        for (int v = 0; v < 8; ++v) {
            int f = kb * 32 + kh * 8 + (v & 3) * 2 + (v >> 2) * 16;
            B.u[v] = pack_bf16(Wp[(size_t)f * D_ + d], Wp[(size_t)(f + 1) * D_ + d]);
        }
        acc = __builtin_amdgcn_wmma_f32_16x16x32_bf16(
            false, A.v, false, B.v, (short)0, acc, false, false);
    }

    uint4 q;
    q.x = pack_bf16(acc[0], acc[1]);
    q.y = pack_bf16(acc[2], acc[3]);
    q.z = pack_bf16(acc[4], acc[5]);
    q.w = pack_bf16(acc[6], acc[7]);
    const int kbk   = r0 >> 5;
    const int khalf = (r0 >> 4) & 1;
    const int fl    = khalf * 16 + n;
    size_t off = ((((size_t)h * 128 + kbk) * 8 + wave) * 32 + fl) * 8 + kh * 4;
    *(uint4*)(whB + off) = q;
}

// ---------------------------------------------------------------------------
// Fused masked-softmax attention + att@Wh (flash style, Sinkhorn eliminated:
// adj>0 == adj0>0 since SK is positive diagonal scaling).
// 8 waves x 16 rows = 128 rows/WG; K streamed in 32-chunks with software-
// pipelined LDS staging. Row sums computed by an extra WMMA against a
// constant all-ones B fragment (rowsum = P @ 1), so no VALU reduction and no
// cross-lane shuffle is needed.
// ---------------------------------------------------------------------------
__global__ __launch_bounds__(256) void attn_kernel(
    const float* __restrict__ adj0, const float* __restrict__ x0,
    const u32* __restrict__ whB, const float* __restrict__ e_src,
    const float* __restrict__ e_dst, float* __restrict__ out) {
    __shared__ __align__(16) float eds[N_];    // 16 KB e_dst row (log2-scaled)
    __shared__ __align__(16) u32   wb[2048];   // 8 KB Wh K-block (B-frag layout)

    const int h     = blockIdx.y;
    const int rbase = blockIdx.x * 128;
    const int tid   = threadIdx.x;
    const int wave  = tid >> 5;
    const int lane  = tid & 31;
    const int m     = lane & 15;
    const int kh    = lane >> 4;
    const int row   = rbase + wave * 16 + m;

    {   // cooperative stage of e_dst[h,:]
        const float* src = e_dst + (size_t)h * N_ + tid * 16;
        #pragma unroll
        for (int i = 0; i < 4; ++i)
            *(float4*)&eds[tid * 16 + i * 4] = *(const float4*)(src + i * 4);
    }
    const float es = e_src[(size_t)h * N_ + row];

    v8f acc[8];
    #pragma unroll
    for (int g = 0; g < 8; ++g) acc[g] = (v8f){};
    v8f acc_s = {};            // rowsum accumulator (P @ ones)

    FragBF ONES;
    #pragma unroll
    for (int v = 0; v < 8; ++v) ONES.u[v] = 0x3F803F80u;  // bf16 1.0 pairs

    const u32*   whBh   = whB + (size_t)h * 128 * 2048;
    const float* adjrow = adj0 + ((size_t)h * N_ + row) * N_;

    // pipeline: preload K-block 0 into registers
    uint4 s0 = *(const uint4*)(whBh + tid * 8);
    uint4 s1 = *(const uint4*)(whBh + tid * 8 + 4);

    for (int kb = 0; kb < 128; ++kb) {
        __syncthreads();                 // readers of previous block done
        *(uint4*)&wb[tid * 8]     = s0;  // publish current block
        *(uint4*)&wb[tid * 8 + 4] = s1;
        __syncthreads();                 // block visible

        if (kb < 127) {                  // prefetch next block into registers
            const u32* gsrc = whBh + (size_t)(kb + 1) * 2048 + tid * 8;
            s0 = *(const uint4*)(gsrc);
            s1 = *(const uint4*)(gsrc + 4);
            __builtin_prefetch(adjrow + (size_t)(kb + 1) * 32 + kh * 8, 0, 0);
        }

        // per-lane mask + e in A-fragment order: row m, K run [kh*16,+16)
        const float* ap = adjrow + kb * 32 + kh * 16;
        const float* ep = &eds[kb * 32 + kh * 16];
        float a[16], ed[16];
        *(float4*)&a[0]   = *(const float4*)(ap);
        *(float4*)&a[4]   = *(const float4*)(ap + 4);
        *(float4*)&a[8]   = *(const float4*)(ap + 8);
        *(float4*)&a[12]  = *(const float4*)(ap + 12);
        *(float4*)&ed[0]  = *(const float4*)(ep);
        *(float4*)&ed[4]  = *(const float4*)(ep + 4);
        *(float4*)&ed[8]  = *(const float4*)(ep + 8);
        *(float4*)&ed[12] = *(const float4*)(ep + 12);

        float p[16];
        #pragma unroll
        for (int i = 0; i < 16; ++i) {
            float e  = es + ed[i];                 // already log2-scaled
            e        = fmaxf(e, ALPHA_ * e);       // leaky relu (scale-commuting)
            float pv = __builtin_amdgcn_exp2f(e);  // raw v_exp_f32
            p[i] = (a[i] > 0.f) ? pv : 0.f;
        }

        FragBF A;
        #pragma unroll
        for (int v = 0; v < 8; ++v) A.u[v] = pack_bf16(p[2 * v], p[2 * v + 1]);

        #pragma unroll
        for (int g = 0; g < 8; ++g) {
            FragBF B;
            const u32* bp = &wb[(g * 32 + lane) * 8];
            *(uint4*)&B.u[0] = *(const uint4*)(bp);
            *(uint4*)&B.u[4] = *(const uint4*)(bp + 4);
            acc[g] = __builtin_amdgcn_wmma_f32_16x16x32_bf16(
                false, A.v, false, B.v, (short)0, acc[g], false, false);
        }
        acc_s = __builtin_amdgcn_wmma_f32_16x16x32_bf16(
            false, A.v, false, ONES.v, (short)0, acc_s, false, false);
    }

    // epilogue: C-layout lane holds rows mm = gg + kh*8; acc_s[gg] is exactly
    // rowsum(mm) for this lane, so normalization is purely lane-local.
    const int halfm = kh * 8;
    #pragma unroll
    for (int g = 0; g < 8; ++g) {
        int dcol = g * 16 + m;
        #pragma unroll
        for (int gg = 0; gg < 8; ++gg) {
            int mm = gg + halfm;
            int r  = rbase + wave * 16 + mm;
            float v = acc[g][gg] / acc_s[gg];
            v = elu_f(v);
            v += x0[(size_t)r * F_ + dcol];
            v = elu_f(v);
            out[(size_t)r * (H_ * D_) + h * D_ + dcol] = v;
        }
    }
}

// ---------------------------------------------------------------------------
extern "C" void kernel_launch(void* const* d_in, const int* in_sizes, int n_in,
                              void* d_out, int out_size, void* d_ws, size_t ws_size,
                              hipStream_t stream) {
    const float* x0    = (const float*)d_in[0];  // [N, F]
    const float* adj0  = (const float*)d_in[1];  // [H, N, N]
    const float* W     = (const float*)d_in[2];  // [H, F, D]
    const float* a_src = (const float*)d_in[3];  // [H, D]
    const float* a_dst = (const float*)d_in[4];  // [H, D]
    float* out = (float*)d_out;                  // [N, H*D]

    char* ws      = (char*)d_ws;
    u32*   whB    = (u32*)ws;                              // 4 MB swizzled Wh
    float* wa_src = (float*)(ws + (size_t)4 * 1024 * 1024);
    float* wa_dst = wa_src + H_ * F_;
    float* e_src  = wa_dst + H_ * F_;
    float* e_dst  = e_src + (size_t)H_ * N_;

    wa_kernel  <<<dim3(4),         dim3(128), 0, stream>>>(W, a_src, a_dst, wa_src, wa_dst);
    e_kernel   <<<dim3(128),       dim3(128), 0, stream>>>(x0, wa_src, wa_dst, e_src, e_dst);
    wh_kernel  <<<dim3(N_/16, H_), dim3(256), 0, stream>>>(x0, W, whB);
    attn_kernel<<<dim3(N_/128, H_),dim3(256), 0, stream>>>(adj0, x0, whB, e_src, e_dst, out);
}
